// HeteroGraphSAGEEncoder_82145544503775
// MI455X (gfx1250) — matrix operand order repaired and checked by reading, not compiled, not run on hardware
//
#include <hip/hip_runtime.h>

// Problem dimensions (match reference)
#define NUSER  200000
#define NMOVIE 50000
#define NTAG   20000
#define FDIM   64
#define HDIM   128
#define HH     (HDIM * HDIM)

typedef __attribute__((ext_vector_type(16))) __bf16 v16bf;
typedef __attribute__((ext_vector_type(8)))  __bf16 v8bf;
typedef __attribute__((ext_vector_type(8)))  float  v8f;

#define AST 136                 // padded LDS stride (bf16 elems) for A tiles
#define WST 136                 // padded LDS stride for transposed weights
#define GEMM_WAVES 4
#define GEMM_ROWS (GEMM_WAVES * 16)

enum { FLAG_ACC = 1, FLAG_ACT = 2 };

// C[n x 128] (+)= bf16(A[n x K] * rowscale) @ bf16(W[128 x K])^T  (+ bias) (then LeakyReLU)
// rowscale = 1/max(cnt[row],1) fuses the segment-mean division into the GEMM.
__global__ __launch_bounds__(GEMM_WAVES * 32) void sage_gemm_bf16(
    const float* __restrict__ A, const float* __restrict__ cnt,
    const float* __restrict__ W, const float* __restrict__ bias,
    float* __restrict__ out, int n_rows, int K, int flags)
{
    __shared__ __bf16 Wt[HDIM * WST];                 // W^T as bf16: Wt[k][n]
    __shared__ __bf16 As[GEMM_WAVES][16 * AST];       // per-wave A tile: As[r][k]

    const int tid  = threadIdx.x;
    const int wave = tid >> 5;
    const int lane = tid & 31;
    const int half = lane >> 4;
    const int lh   = lane & 15;
    const int K4   = K >> 2;

    // Stage W^T into LDS (coalesced f32 reads along K, bf16 convert on store)
    for (int i = tid; i < HDIM * K4; i += GEMM_WAVES * 32) {
        const int n  = i / K4;
        const int kc = (i - n * K4) << 2;
        const float4 w4 = *(const float4*)(W + (size_t)n * K + kc);
        Wt[(kc + 0) * WST + n] = (__bf16)w4.x;
        Wt[(kc + 1) * WST + n] = (__bf16)w4.y;
        Wt[(kc + 2) * WST + n] = (__bf16)w4.z;
        Wt[(kc + 3) * WST + n] = (__bf16)w4.w;
    }

    // Stage this wave's 16 x K A tile (scaled by 1/max(cnt,1) if cnt given)
    const int rowBase = blockIdx.x * GEMM_ROWS + wave * 16;
    __bf16* Aw = &As[wave][0];
    for (int i = lane; i < 16 * K4; i += 32) {
        const int r = i / K4;
        const int c = (i - r * K4) << 2;
        const int row = rowBase + r;
        float4 a4 = {0.f, 0.f, 0.f, 0.f};
        float s = 1.0f;
        if (row < n_rows) {
            a4 = *(const float4*)(A + (size_t)row * K + c);
            if (cnt) s = 1.0f / fmaxf(cnt[row], 1.0f);
        }
        Aw[r * AST + c + 0] = (__bf16)(a4.x * s);
        Aw[r * AST + c + 1] = (__bf16)(a4.y * s);
        Aw[r * AST + c + 2] = (__bf16)(a4.z * s);
        Aw[r * AST + c + 3] = (__bf16)(a4.w * s);
    }
    __syncthreads();

    v8f acc[8] = {};

    for (int kb = 0; kb < K; kb += 32) {
        // A fragment per ISA layout: lanes 0-15 -> K{0..7,16..23}, lanes 16-31 -> K{8..15,24..31}
        const v8bf* ar = (const v8bf*)(Aw + lh * AST + kb);
        const v16bf afrag = __builtin_shufflevector(ar[half], ar[2 + half],
            0,1,2,3,4,5,6,7,8,9,10,11,12,13,14,15);
        // B fragment: lane <-> N column of W^T, K{0..7,16..23} / K{8..15,24..31} per half
        const __bf16* brow = Wt + (kb + 8 * half) * WST + lh;
#pragma unroll
        for (int nt = 0; nt < 8; ++nt) {
            v16bf bfrag;
#pragma unroll
            for (int s = 0; s < 16; ++s)
                bfrag[s] = brow[(s < 8 ? s : s + 8) * WST + nt * 16];
            acc[nt] = __builtin_amdgcn_wmma_f32_16x16x32_bf16(
                false, afrag, false, bfrag, (short)0, acc[nt], false, false);
        }
    }

    // Epilogue: C layout lanes 0-15 N=lane / M=r, lanes 16-31 N=lane-16 / M=8+r
#pragma unroll
    for (int nt = 0; nt < 8; ++nt) {
        const int n = nt * 16 + lh;
        const float bv = bias ? bias[n] : 0.0f;
#pragma unroll
        for (int r = 0; r < 8; ++r) {
            const int row = rowBase + r + 8 * half;
            if (row < n_rows) {
                float v = acc[nt][r] + bv;
                const size_t idx = (size_t)row * HDIM + n;
                if (flags & FLAG_ACC) v += out[idx];
                if (flags & FLAG_ACT) v = (v > 0.f) ? v : 0.1f * v;
                out[idx] = v;
            }
        }
    }
}

// One wave per edge: gather src row (coalesced float4), scatter-add to dst row + count.
__global__ __launch_bounds__(256) void edge_scatter(
    const float* __restrict__ hsrc, const int* __restrict__ src,
    const int* __restrict__ dst, float* __restrict__ accum,
    float* __restrict__ cnt, int nE)
{
    const int e = blockIdx.x * 8 + (threadIdx.x >> 5);
    if (e >= nE) return;
    const int lane = threadIdx.x & 31;
    const int s = src[e];
    const int d = dst[e];
    const float4 v = *(const float4*)(hsrc + (size_t)s * HDIM + lane * 4);
    float* o = accum + (size_t)d * HDIM + lane * 4;
    unsafeAtomicAdd(o + 0, v.x);
    unsafeAtomicAdd(o + 1, v.y);
    unsafeAtomicAdd(o + 2, v.z);
    unsafeAtomicAdd(o + 3, v.w);
    if (lane == 0) unsafeAtomicAdd(cnt + d, 1.0f);
}

extern "C" void kernel_launch(void* const* d_in, const int* in_sizes, int n_in,
                              void* d_out, int out_size, void* d_ws, size_t ws_size,
                              hipStream_t stream)
{
    (void)n_in; (void)out_size; (void)ws_size;

    const float* genre    = (const float*)d_in[0];
    const float* user_emb = (const float*)d_in[1];
    const float* tag_emb  = (const float*)d_in[2];
    const float* movie_W  = (const float*)d_in[3];
    const float* movie_b  = (const float*)d_in[4];
    const float* Wn       = (const float*)d_in[5];   // [L,4,H,H]
    const float* Ws       = (const float*)d_in[6];   // [L,4,H,H]
    const float* bs       = (const float*)d_in[7];   // [L,4,H]
    const int*   rates_u  = (const int*)d_in[8];
    const int*   rates_m  = (const int*)d_in[9];
    const int*   tag_m    = (const int*)d_in[10];
    const int*   tag_t    = (const int*)d_in[11];
    const int E1 = in_sizes[8];
    const int E2 = in_sizes[10];

    // Workspace carve
    char* p = (char*)d_ws;
    auto carve = [&](size_t bytes) -> float* {
        float* r = (float*)p;
        p += (bytes + 255) & ~(size_t)255;
        return r;
    };
    float* hm0  = carve((size_t)NMOVIE * HDIM * 4);
    float* hu1  = carve((size_t)NUSER  * HDIM * 4);
    float* hm1  = carve((size_t)NMOVIE * HDIM * 4);
    float* ht1  = carve((size_t)NTAG   * HDIM * 4);
    float* mean = carve((size_t)NUSER  * HDIM * 4);  // shared scratch, max-sized
    float* cntb = carve((size_t)NUSER  * 4);

    auto gemm = [&](const float* A, const float* cnt, const float* W,
                    const float* bias, float* out, int n, int K, int flags) {
        dim3 g((n + GEMM_ROWS - 1) / GEMM_ROWS);
        hipLaunchKernelGGL(sage_gemm_bf16, g, dim3(GEMM_WAVES * 32), 0, stream,
                           A, cnt, W, bias, out, n, K, flags);
    };
    auto scatter = [&](const float* hsrc, const int* src, const int* dst,
                       int nDst, int nE) {
        hipMemsetAsync(mean, 0, (size_t)nDst * HDIM * 4, stream);
        hipMemsetAsync(cntb, 0, (size_t)nDst * 4, stream);
        hipLaunchKernelGGL(edge_scatter, dim3((nE + 7) / 8), dim3(256), 0, stream,
                           hsrc, src, dst, mean, cntb, nE);
    };

    // Movie input projection: hm0 = genre @ movie_W^T + movie_b   (K = 64)
    gemm(genre, nullptr, movie_W, movie_b, hm0, NMOVIE, FDIM, 0);

    const float* hu = user_emb;
    const float* hm = hm0;
    const float* ht = tag_emb;

    float* out_hu = (float*)d_out;                       // [NUSER, H]
    float* out_hm = (float*)d_out + (size_t)NUSER * HDIM; // [NMOVIE, H]

    // etype order: 0=rates(u->m), 1=rev_rates(m->u), 2=has_tag(m->t), 3=tag_of(t->m)
    for (int l = 0; l < 2; ++l) {
        const bool last = (l == 1);
        float* nm = last ? out_hm : hm1;
        float* nu = last ? out_hu : hu1;
        const float* Wn_l = Wn + (size_t)l * 4 * HH;
        const float* Ws_l = Ws + (size_t)l * 4 * HH;
        const float* bs_l = bs + (size_t)l * 4 * HDIM;

        // movie = sage(rates) + sage(tag_of), LeakyReLU on final accumulation
        gemm(hm, nullptr, Ws_l + 0 * HH, bs_l + 0 * HDIM, nm, NMOVIE, HDIM, 0);
        gemm(hm, nullptr, Ws_l + 3 * HH, bs_l + 3 * HDIM, nm, NMOVIE, HDIM, FLAG_ACC);
        scatter(hu, rates_u, rates_m, NMOVIE, E1);
        gemm(mean, cntb, Wn_l + 0 * HH, nullptr, nm, NMOVIE, HDIM, FLAG_ACC);
        scatter(ht, tag_t, tag_m, NMOVIE, E2);
        gemm(mean, cntb, Wn_l + 3 * HH, nullptr, nm, NMOVIE, HDIM, FLAG_ACC | FLAG_ACT);

        // user = sage(rev_rates)
        gemm(hu, nullptr, Ws_l + 1 * HH, bs_l + 1 * HDIM, nu, NUSER, HDIM, 0);
        scatter(hm, rates_m, rates_u, NUSER, E1);
        gemm(mean, cntb, Wn_l + 1 * HH, nullptr, nu, NUSER, HDIM, FLAG_ACC | FLAG_ACT);

        if (!last) {
            // tag = sage(has_tag) — only needed as input to layer 1
            gemm(ht, nullptr, Ws_l + 2 * HH, bs_l + 2 * HDIM, ht1, NTAG, HDIM, 0);
            scatter(hm, tag_m, tag_t, NTAG, E2);
            gemm(mean, cntb, Wn_l + 2 * HH, nullptr, ht1, NTAG, HDIM, FLAG_ACC | FLAG_ACT);
            ht = ht1;
        }
        hu = nu;
        hm = nm;
    }
}
